// QuantizedLinear_30623116820582
// MI455X (gfx1250) — matrix operand order repaired and checked by reading, not compiled, not run on hardware
//
#include <hip/hip_runtime.h>
#include <stdint.h>

// QuantizedLinear: Y[M,N] = (X[M,K] @ Wq[N,K]^T) * scale + bias
// M = 8192, N = 16384, K = 4096.
// Fast path: one-pass convert (X->f16, Wq->f16; int8 exact in f16) into d_ws,
// then pure f16 WMMA GEMM with GLOBAL_LOAD_ASYNC_TO_LDS_B128 LDS fills.
// Buffer/matrix selection uses the async-load property that INST_OFFSET is
// added to BOTH the LDS and global addresses (ISA 08_async_tensor §4.4):
// LDS side selected by imm offset, global side compensated in the SGPR base.
// Fallback (small ws): fused dequant GEMM.

typedef _Float16 h4   __attribute__((ext_vector_type(4)));
typedef _Float16 h8   __attribute__((ext_vector_type(8)));
typedef _Float16 v16h __attribute__((ext_vector_type(16)));
typedef float    v8f  __attribute__((ext_vector_type(8)));

#define M_TOT 8192
#define N_TOT 16384
#define K_TOT 4096
#define BM 128
#define BN 128
#define BK 64
#define LDK (BK + 8)                  // +16B row pad: conflict-free LDS reads
#define NKT (K_TOT / BK)              // 64 K-steps
#define TILE_H (BM * LDK)             // halves per tile buffer
#define TILE_B (TILE_H * 2)           // 18432 bytes per tile buffer

// ---------------------------------------------------------------------------
// conversion kernels (bandwidth-bound, one pass)
// ---------------------------------------------------------------------------
__global__ __launch_bounds__(256)
void cvt_x_kernel(const float* __restrict__ X, _Float16* __restrict__ Xh)
{
    const size_t i = ((size_t)blockIdx.x * 256 + threadIdx.x) * 8;
    float4 f0 = *(const float4*)(X + i);
    float4 f1 = *(const float4*)(X + i + 4);
    h8 h;
    h[0] = (_Float16)f0.x; h[1] = (_Float16)f0.y;
    h[2] = (_Float16)f0.z; h[3] = (_Float16)f0.w;
    h[4] = (_Float16)f1.x; h[5] = (_Float16)f1.y;
    h[6] = (_Float16)f1.z; h[7] = (_Float16)f1.w;
    *(h8*)(Xh + i) = h;
}

__global__ __launch_bounds__(256)
void cvt_w_kernel(const int8_t* __restrict__ W, _Float16* __restrict__ Wh)
{
    const size_t i = ((size_t)blockIdx.x * 256 + threadIdx.x) * 16;
    int4 v = *(const int4*)(W + i);
    const int* dw = (const int*)&v;
    h8 lo, hi;
    #pragma unroll
    for (int e = 0; e < 8; ++e) {
        lo[e] = (_Float16)(int8_t)(dw[e >> 2]       >> (8 * (e & 3)));
        hi[e] = (_Float16)(int8_t)(dw[(e >> 2) + 2] >> (8 * (e & 3)));
    }
    *(h8*)(Wh + i)     = lo;
    *(h8*)(Wh + i + 8) = hi;
}

// ---------------------------------------------------------------------------
// fast path: f16 GEMM, async LDS fills, 4 waves x (64x64) per 128x128 block
// ---------------------------------------------------------------------------
__device__ __forceinline__ void wait_async0()
{
    asm volatile("s_wait_asynccnt 0x0" ::: "memory");
}

// Issue one 128x64-f16 A tile + one 128x64-f16 B tile as 16 async b128 DMAs
// per thread-group slice. LDS destination VGPRs are shared by all four
// (matrix, buf) combinations; selection is a compile-time imm offset whose
// global-side contribution is cancelled in the SGPR base.
template <int BUF>
__device__ __forceinline__ void issue_tile(const uint32_t (&lds0)[8],
                                           const uint32_t (&goff)[8],
                                           uint64_t baseA, uint64_t baseB,
                                           int kt)
{
    constexpr int offA = BUF * TILE_B;            // A buf: 0 / 18432
    constexpr int offB = (2 + BUF) * TILE_B;      // B buf: 36864 / 55296
    const uint64_t sa = baseA + (uint64_t)kt * (BK * 2) - (uint64_t)offA;
    const uint64_t sb = baseB + (uint64_t)kt * (BK * 2) - (uint64_t)offB;
    #pragma unroll
    for (int j = 0; j < 8; ++j)
        asm volatile("global_load_async_to_lds_b128 %0, %1, %2 offset:%3"
                     :: "v"(lds0[j]), "v"(goff[j]), "s"(sa), "n"(offA)
                     : "memory");
    #pragma unroll
    for (int j = 0; j < 8; ++j)
        asm volatile("global_load_async_to_lds_b128 %0, %1, %2 offset:%3"
                     :: "v"(lds0[j]), "v"(goff[j]), "s"(sb), "n"(offB)
                     : "memory");
}

__global__ __launch_bounds__(128)
void qlinear_wmma_async(const _Float16* __restrict__ Xh,
                        const _Float16* __restrict__ Wh,
                        const float* __restrict__ scale,
                        const float* __restrict__ bias,
                        float* __restrict__ Y)
{
    // one arena: [A buf0 | A buf1 | B buf0 | B buf1], each BM x LDK halves
    __shared__ _Float16 smem[4 * TILE_H];

    const int tid  = threadIdx.x;
    const int lane = tid & 31;
    const int wave = tid >> 5;        // 0..3
    const int wm   = wave & 1;        // 2 waves over M (64 rows each)
    const int wn   = wave >> 1;       // 2 waves over N (64 cols each)
    const int m0   = blockIdx.y * BM;
    const int n0   = blockIdx.x * BN;

    // Fill distribution: tile = 128 rows x 8 chunks of 16B = 1024 chunks,
    // 128 threads -> 8 chunks/thread (rows f_row + 16j, fixed 16B segment).
    const int f_row = tid >> 3;       // 0..15
    const int f_seg = tid & 7;        // 16B chunk within row

    uint32_t lds0[8];                 // LDS byte address of A-buf0 slot
    uint32_t goff[8];                 // global byte offset within operand
    #pragma unroll
    for (int j = 0; j < 8; ++j) {
        const int row = f_row + 16 * j;
        lds0[j] = (uint32_t)(uintptr_t)&smem[row * LDK + f_seg * 8];
        goff[j] = (uint32_t)(row * (K_TOT * 2) + f_seg * 16);
    }
    const uint64_t baseA = (uint64_t)(uintptr_t)(Xh + (size_t)m0 * K_TOT);
    const uint64_t baseB = (uint64_t)(uintptr_t)(Wh + (size_t)n0 * K_TOT);

    v8f acc[4][4] = {};
    const int rl = lane & 15;         // row/col within 16-tile
    const int ch = lane >> 4;         // half-wave select

    auto compute = [&](int buf) {
        const _Float16* Ab = smem + buf * TILE_H;
        const _Float16* Bb = smem + (2 + buf) * TILE_H;
        #pragma unroll
        for (int kk = 0; kk < BK; kk += 32) {
            v16h a[4], b[4];
            // A 16x32 f16: lanes 0-15 hold K {0..7,16..23}, lanes 16-31 hold
            // K {8..15,24..31} for row M=lane&15 (ISA 7.12.2).
            #pragma unroll
            for (int mt = 0; mt < 4; ++mt) {
                const _Float16* p = Ab + (wm * 64 + mt * 16 + rl) * LDK
                                       + kk + ch * 8;
                h8 p0 = *(const h8*)(p);
                h8 p1 = *(const h8*)(p + 16);
                a[mt] = __builtin_shufflevector(p0, p1,
                        0,1,2,3,4,5,6,7, 8,9,10,11,12,13,14,15);
            }
            // B 32x16 f16: lanes 0-15 hold K {0..7,16..23}, lanes 16-31 hold
            // K {8..15,24..31} for column N=lane&15 (same order as A).
            #pragma unroll
            for (int nt = 0; nt < 4; ++nt) {
                const _Float16* p = Bb + (wn * 64 + nt * 16 + rl) * LDK
                                       + kk + ch * 8;
                h8 p0 = *(const h8*)(p);
                h8 p1 = *(const h8*)(p + 16);
                b[nt] = __builtin_shufflevector(p0, p1,
                        0,1,2,3,4,5,6,7, 8,9,10,11,12,13,14,15);
            }
            #pragma unroll
            for (int mt = 0; mt < 4; ++mt)
                #pragma unroll
                for (int nt = 0; nt < 4; ++nt)
                    acc[mt][nt] = __builtin_amdgcn_wmma_f32_16x16x32_f16(
                        false, a[mt], false, b[nt],
                        (short)0, acc[mt][nt], false, false);
        }
    };

    issue_tile<0>(lds0, goff, baseA, baseB, 0);
    wait_async0();
    __syncthreads();

    #pragma unroll 1
    for (int kt = 0; kt < NKT; kt += 2) {          // NKT even
        issue_tile<1>(lds0, goff, baseA, baseB, kt + 1);  // DMA during WMMA
        compute(0);
        wait_async0();
        __syncthreads();
        if (kt + 2 < NKT)
            issue_tile<0>(lds0, goff, baseA, baseB, kt + 2);
        compute(1);
        if (kt + 2 < NKT) {
            wait_async0();
            __syncthreads();
        }
    }

    // epilogue: y = acc*scale + bias  (C layout: VGPR i -> row i+8*ch, col rl)
    const float s = scale[0];
    #pragma unroll
    for (int nt = 0; nt < 4; ++nt) {
        const int n    = n0 + wn * 64 + nt * 16 + rl;
        const float bn = bias[n];
        #pragma unroll
        for (int mt = 0; mt < 4; ++mt) {
            const int mbase = m0 + wm * 64 + mt * 16 + ch * 8;
            #pragma unroll
            for (int i = 0; i < 8; ++i)
                Y[(size_t)(mbase + i) * N_TOT + n] = acc[mt][nt][i] * s + bn;
        }
    }
}

// ---------------------------------------------------------------------------
// fallback: fused dequant GEMM, used when ws is too small
// ---------------------------------------------------------------------------
__global__ __launch_bounds__(256)
void qlinear_wmma_fused(const float* __restrict__ X,
                        const int8_t* __restrict__ W,
                        const float* __restrict__ scale,
                        const float* __restrict__ bias,
                        float* __restrict__ Y)
{
    __shared__ _Float16 As[2][BM][LDK];
    __shared__ _Float16 Bs[2][BN][LDK];

    const int tid  = threadIdx.x;
    const int lane = tid & 31;
    const int wave = tid >> 5;
    const int wm   = wave & 3;
    const int wn   = wave >> 2;
    const int m0 = blockIdx.y * BM;
    const int n0 = blockIdx.x * BN;

    const int a_c = (tid & 15) * 4;
    const int a_r = tid >> 4;
    const int b_r = tid >> 1;
    const int b_c = (tid & 1) * 32;

    float4 aReg[8];
    int4   bReg[2];

    auto loadG = [&](int kt) {
        const float* ap = X + (size_t)m0 * K_TOT + (size_t)kt * BK + a_c;
        #pragma unroll
        for (int r = 0; r < 8; ++r)
            aReg[r] = *(const float4*)(ap + (size_t)(a_r + 16 * r) * K_TOT);
        const int8_t* bp = W + (size_t)(n0 + b_r) * K_TOT + (size_t)kt * BK + b_c;
        bReg[0] = *(const int4*)(bp);
        bReg[1] = *(const int4*)(bp + 16);
    };

    auto storeL = [&](int buf) {
        #pragma unroll
        for (int r = 0; r < 8; ++r) {
            h4 h;
            h[0] = (_Float16)aReg[r].x;
            h[1] = (_Float16)aReg[r].y;
            h[2] = (_Float16)aReg[r].z;
            h[3] = (_Float16)aReg[r].w;
            *(h4*)&As[buf][a_r + 16 * r][a_c] = h;
        }
        #pragma unroll
        for (int s = 0; s < 2; ++s) {
            const int* dw = (const int*)&bReg[s];
            h8 lo, hi;
            #pragma unroll
            for (int e = 0; e < 8; ++e) {
                lo[e] = (_Float16)(int8_t)(dw[e >> 2]       >> (8 * (e & 3)));
                hi[e] = (_Float16)(int8_t)(dw[(e >> 2) + 2] >> (8 * (e & 3)));
            }
            *(h8*)&Bs[buf][b_r][b_c + s * 16]     = lo;
            *(h8*)&Bs[buf][b_r][b_c + s * 16 + 8] = hi;
        }
    };

    v8f acc[2][4] = {};
    const int rl = lane & 15;
    const int ch = lane >> 4;

    auto compute = [&](int buf) {
        #pragma unroll
        for (int kk = 0; kk < BK; kk += 32) {
            v16h a[2], b[4];
            #pragma unroll
            for (int mt = 0; mt < 2; ++mt) {
                const _Float16* p = &As[buf][wm * 32 + mt * 16 + rl][kk + ch * 8];
                h8 p0 = *(const h8*)(p);
                h8 p1 = *(const h8*)(p + 16);
                a[mt] = __builtin_shufflevector(p0, p1,
                        0,1,2,3,4,5,6,7, 8,9,10,11,12,13,14,15);
            }
            #pragma unroll
            for (int nt = 0; nt < 4; ++nt) {
                const _Float16* p = &Bs[buf][wn * 64 + nt * 16 + rl][kk + ch * 8];
                h8 p0 = *(const h8*)(p);
                h8 p1 = *(const h8*)(p + 16);
                b[nt] = __builtin_shufflevector(p0, p1,
                        0,1,2,3,4,5,6,7, 8,9,10,11,12,13,14,15);
            }
            #pragma unroll
            for (int mt = 0; mt < 2; ++mt)
                #pragma unroll
                for (int nt = 0; nt < 4; ++nt)
                    acc[mt][nt] = __builtin_amdgcn_wmma_f32_16x16x32_f16(
                        false, a[mt], false, b[nt],
                        (short)0, acc[mt][nt], false, false);
        }
    };

    loadG(0);
    storeL(0);
    __syncthreads();

    for (int kt = 0; kt < NKT; ++kt) {
        const int buf = kt & 1;
        if (kt + 1 < NKT) loadG(kt + 1);
        compute(buf);
        if (kt + 1 < NKT) {
            storeL(buf ^ 1);
            __syncthreads();
        }
    }

    const float s = scale[0];
    #pragma unroll
    for (int nt = 0; nt < 4; ++nt) {
        const int n    = n0 + wn * 64 + nt * 16 + rl;
        const float bn = bias[n];
        #pragma unroll
        for (int mt = 0; mt < 2; ++mt) {
            const int mbase = m0 + wm * 32 + mt * 16 + ch * 8;
            #pragma unroll
            for (int i = 0; i < 8; ++i)
                Y[(size_t)(mbase + i) * N_TOT + n] = acc[mt][nt][i] * s + bn;
        }
    }
}

// ---------------------------------------------------------------------------
extern "C" void kernel_launch(void* const* d_in, const int* in_sizes, int n_in,
                              void* d_out, int out_size, void* d_ws, size_t ws_size,
                              hipStream_t stream) {
    const float*  X     = (const float*)d_in[0];
    const int8_t* W     = (const int8_t*)d_in[1];
    const float*  scale = (const float*)d_in[2];
    const float*  bias  = (const float*)d_in[3];
    float*        Y     = (float*)d_out;

    const size_t needA = (size_t)M_TOT * K_TOT * 2;   // f16 X
    const size_t needB = (size_t)N_TOT * K_TOT * 2;   // f16 W
    dim3 grid(N_TOT / BN, M_TOT / BM);                // 128 x 64 workgroups

    if (ws_size >= needA + needB) {
        _Float16* Xh = (_Float16*)d_ws;
        _Float16* Wh = (_Float16*)((char*)d_ws + needA);
        cvt_x_kernel<<<(M_TOT * (size_t)K_TOT) / (256 * 8), 256, 0, stream>>>(X, Xh);
        cvt_w_kernel<<<(N_TOT * (size_t)K_TOT) / (256 * 16), 256, 0, stream>>>(W, Wh);
        qlinear_wmma_async<<<grid, 128, 0, stream>>>(Xh, Wh, scale, bias, Y);
    } else {
        qlinear_wmma_fused<<<grid, 256, 0, stream>>>(X, W, scale, bias, Y);
    }
}
